// MultiHeadAttention_37134287242046
// MI455X (gfx1250) — matrix operand order repaired and checked by reading, not compiled, not run on hardware
//
#include <hip/hip_runtime.h>
#include <hip/hip_bf16.h>

// ---------------------------------------------------------------------------
// MHA forward for MI455X (gfx1250): f16 WMMA (v_wmma_f32_16x16x32_f16), fp32
// accumulate.  MI455X-specific design points:
//  * 192 MB L2 -> GEMMs feed WMMA fragments DIRECTLY from global/L2 (no LDS,
//    no barriers): A is row-major f16 [M,K] (contiguous A-fragments) and W is
//    pre-transposed to f16 [N,K] (contiguous B-fragments).
//  * Cache policy: the one-shot 536 MB fp32 bias stream uses non-temporal
//    loads (TH=NT) so it cannot evict the reused f16 working set from L2;
//    the final fp32 output uses non-temporal stores.  Reused streams (A, K
//    rows) are prefetched at WGP scope (locality 3).
//  * wave32 flash attention: one wave owns a 16-row Q tile, online softmax
//    with 16-lane __shfl_xor reductions, P lane-layout transposed through a
//    private LDS tile.
// Pipeline:
//   0) cvt_f16        : q,k,v fp32 -> f16 [M,1024]
//   0) cvt_tr_f16     : w_q,w_k,w_v,w_o fp32 [K,N] -> f16 [N,K]
//   1) gemm<HEAD>     : Qh = (q@w_q + b_q) * 0.125 -> f16 [B,H,S,64]
//   2) gemm<HEAD>     : Kh = (k@w_k + b_k)         -> f16 [B,H,S,64]
//   3) gemm<HEAD_T>   : Vt = (v@w_v + b_v)^T       -> f16 [B,H,64,S]
//   4) flash_attn     : online-softmax attention   -> f16 [B,S,1024]
//   5) gemm<FINAL>    : out = attn@w_o + b_o       -> f32 [B,S,1024]
// ---------------------------------------------------------------------------

typedef _Float16 v16h __attribute__((ext_vector_type(16)));
typedef _Float16 v8h  __attribute__((ext_vector_type(8)));
typedef float    v8f  __attribute__((ext_vector_type(8)));

#define WMMA_F16(a, b, c) \
  __builtin_amdgcn_wmma_f32_16x16x32_f16(false, (a), false, (b), (short)0, (c), false, false)

static constexpr int BATCH = 2;
static constexpr int SEQ   = 2048;
static constexpr int HEADS = 16;
static constexpr int DHEAD = 64;
static constexpr int DM    = 1024;          // d_model
static constexpr int MROWS = BATCH * SEQ;   // 4096

enum OutMode { OUT_HEAD = 0, OUT_HEAD_T = 1, OUT_FINAL = 2 };

// ---------------------------------------------------------------------------
// fp32 -> f16 pack convert (4 elements / thread, vectorized b128 load)
// ---------------------------------------------------------------------------
__global__ __launch_bounds__(256) void cvt_f16(const float* __restrict__ in,
                                               _Float16* __restrict__ out, int n) {
  const int i = (blockIdx.x * 256 + threadIdx.x) * 4;
  if (i + 3 < n) {
#pragma unroll
    for (int j = 0; j < 4; ++j) out[i + j] = (_Float16)in[i + j];
  }
}

// ---------------------------------------------------------------------------
// fp32 [K,N] -> f16 [N,K] transpose-convert via LDS 32x32 tile
// ---------------------------------------------------------------------------
__global__ __launch_bounds__(256) void cvt_tr_f16(const float* __restrict__ W,
                                                  _Float16* __restrict__ WT) {
  __shared__ _Float16 tile[32][33];
  const int tid = threadIdx.x;
  const int k0 = blockIdx.y * 32, n0 = blockIdx.x * 32;
  const int r  = tid >> 3;        // 0..31
  const int c4 = (tid & 7) * 4;   // 0..28
#pragma unroll
  for (int j = 0; j < 4; ++j)
    tile[r][c4 + j] = (_Float16)W[(size_t)(k0 + r) * DM + n0 + c4 + j];
  __syncthreads();
#pragma unroll
  for (int j = 0; j < 4; ++j)
    WT[(size_t)(n0 + r) * DM + k0 + c4 + j] = tile[c4 + j][r];
}

// ---------------------------------------------------------------------------
// LDS-free f16 WMMA GEMM: out = A[M,K] @ W[K,N] + bias[N] (M=4096, N=K=1024)
// A row-major f16; W given pre-transposed WT[N,K] f16.  All fragments are
// contiguous per-lane global loads served from the 192 MB L2.
// Block = 8 waves; wave tile 16x64; block tile 128(M) x 64(N).
// ---------------------------------------------------------------------------
template <int MODE>
__global__ __launch_bounds__(256) void gemm_wmma(const _Float16* __restrict__ A,
                                                 const _Float16* __restrict__ WT,
                                                 const float* __restrict__ bias,
                                                 void* __restrict__ outp,
                                                 float scale) {
  const int tid  = threadIdx.x;
  const int wid  = tid >> 5;
  const int lane = tid & 31;
  const int lr   = lane & 15;   // A row / B-C column within 16 (ISA layouts)
  const int lh   = lane >> 4;   // half-wave select
  const int m0 = blockIdx.y * 128 + wid * 16;
  const int n0 = blockIdx.x * 64;

  v8f acc[4];
#pragma unroll
  for (int t = 0; t < 4; ++t)
#pragma unroll
    for (int i = 0; i < 8; ++i) acc[t][i] = 0.0f;

  const _Float16* arow = A + (size_t)(m0 + lr) * DM;
  const int kb  = lh * 16;   // A frag: elem j = K kb+j
  const int kbb = lh * 16;   // B frag: elem j = K kbb+j

#pragma unroll 4
  for (int k0 = 0; k0 < DM; k0 += 32) {
    union { v16h v; v8h p[2]; } a;
    a.p[0] = *(const v8h*)(arow + k0 + kb);
    a.p[1] = *(const v8h*)(arow + k0 + kb + 8);
    __builtin_prefetch(arow + k0 + 128, 0, 3);  // global_prefetch_b8, WGP scope
#pragma unroll
    for (int t = 0; t < 4; ++t) {
      const _Float16* wr = WT + (size_t)(n0 + t * 16 + lr) * DM + k0 + kbb;
      v16h bf = *(const v16h*)wr;
      acc[t] = WMMA_F16(a.v, bf, acc[t]);
    }
  }

  // epilogue: C layout -> row = (lane/16)*8 + i, col = lane%16 (+ tile offs)
  const int grow0 = m0 + lh * 8;
#pragma unroll
  for (int t = 0; t < 4; ++t) {
    const int col = n0 + t * 16 + lr;
    const float bv = bias[col];
#pragma unroll
    for (int i = 0; i < 8; ++i) {
      const int row = grow0 + i;
      const float x = (acc[t][i] + bv) * scale;
      if (MODE == OUT_FINAL) {
        // final output is never re-read: non-temporal store (TH=NT)
        __builtin_nontemporal_store(x, &((float*)outp)[(size_t)row * DM + col]);
      } else {
        const int b = row / SEQ, s = row % SEQ;
        const int hh = col >> 6, dd = col & 63;
        _Float16* oh = (_Float16*)outp;
        if (MODE == OUT_HEAD)
          oh[((size_t)(b * HEADS + hh) * SEQ + s) * DHEAD + dd] = (_Float16)x;
        else  // OUT_HEAD_T: [B,H,64,S]
          oh[((size_t)(b * HEADS + hh) * DHEAD + dd) * SEQ + s] = (_Float16)x;
      }
    }
  }
}

// ---------------------------------------------------------------------------
// Flash attention: one wave owns a 16-row Q tile (all 64 dims), streams keys
// in chunks of 32 with online softmax; fp32 bias added to fp32 WMMA scores.
// P is round-tripped through a private LDS tile to convert the score C-layout
// (lane = column) into the A-fragment layout (lane = row) for the PV matmul.
// ---------------------------------------------------------------------------
__global__ __launch_bounds__(256) void flash_attn(const _Float16* __restrict__ Qh,
                                                  const _Float16* __restrict__ Kh,
                                                  const _Float16* __restrict__ Vt,
                                                  const float* __restrict__ bias,
                                                  _Float16* __restrict__ Out) {
  __shared__ _Float16 Plds[8][16][32];  // per-wave 16x32 P tile, 8 KB total

  const int tid  = threadIdx.x;
  const int wid  = tid >> 5;
  const int lane = tid & 31;
  const int lr   = lane & 15;
  const int lh   = lane >> 4;
  const int b = blockIdx.z, h = blockIdx.y;
  const int qbase = blockIdx.x * 128 + wid * 16;

  const size_t hoff = (size_t)(b * HEADS + h);
  const _Float16* Qp = Qh + hoff * SEQ * DHEAD;
  const _Float16* Kp = Kh + hoff * SEQ * DHEAD;
  const _Float16* Vp = Vt + hoff * DHEAD * SEQ;
  const size_t bbase = (hoff * SEQ + qbase) * SEQ;

  // Q fragments for d=0..31 and d=32..63 (1/sqrt(64) already folded into Qh)
  union U { v16h v; v8h p[2]; };
  U qa0, qa1;
  {
    const _Float16* qr = Qp + (size_t)(qbase + lr) * DHEAD;
    const int kb = lh * 16;
    qa0.p[0] = *(const v8h*)(qr + kb);
    qa0.p[1] = *(const v8h*)(qr + kb + 8);
    qa1.p[0] = *(const v8h*)(qr + 32 + kb);
    qa1.p[1] = *(const v8h*)(qr + 32 + kb + 8);
  }

  v8f o[4];
  float mrow[8], lrow[8];
#pragma unroll
  for (int t = 0; t < 4; ++t)
#pragma unroll
    for (int i = 0; i < 8; ++i) o[t][i] = 0.0f;
#pragma unroll
  for (int i = 0; i < 8; ++i) { mrow[i] = -1e30f; lrow[i] = 0.0f; }

  const int db = lh * 16;
  for (int k0 = 0; k0 < SEQ; k0 += 32) {
    // K^T B-fragments: contiguous 32B rows of Kh (lane = key column)
    const _Float16* kr0 = Kp + (size_t)(k0 + lr) * DHEAD;
    const _Float16* kr1 = kr0 + 16 * DHEAD;
    __builtin_prefetch(kr0 + 32 * DHEAD, 0, 3);  // next chunk K rows, WGP scope
    v16h kb0a = *(const v16h*)(kr0 + db);
    v16h kb0b = *(const v16h*)(kr0 + 32 + db);
    v16h kb1a = *(const v16h*)(kr1 + db);
    v16h kb1b = *(const v16h*)(kr1 + 32 + db);

    v8f s0, s1;
#pragma unroll
    for (int i = 0; i < 8; ++i) { s0[i] = 0.0f; s1[i] = 0.0f; }
    s0 = WMMA_F16(qa0.v, kb0a, s0);
    s0 = WMMA_F16(qa1.v, kb0b, s0);
    s1 = WMMA_F16(qa0.v, kb1a, s1);
    s1 = WMMA_F16(qa1.v, kb1b, s1);

#pragma unroll
    for (int i = 0; i < 8; ++i) {
      const int r = lh * 8 + i;  // row within the 16-row tile
      const size_t bro = bbase + (size_t)r * SEQ + k0 + lr;
      // bias is a one-shot 536 MB stream: non-temporal loads keep it from
      // evicting the reused f16 Q/K/V working set from the 192 MB L2
      float x0 = s0[i] + __builtin_nontemporal_load(bias + bro);
      float x1 = s1[i] + __builtin_nontemporal_load(bias + bro + 16);
      // row max across the 16 lanes holding this row's 32 columns
      float mx = fmaxf(x0, x1);
#pragma unroll
      for (int off = 8; off >= 1; off >>= 1) mx = fmaxf(mx, __shfl_xor(mx, off, 16));
      const float mn = fmaxf(mrow[i], mx);
      const float alpha = __expf(mrow[i] - mn);
      mrow[i] = mn;
      const float p0 = __expf(x0 - mn);
      const float p1 = __expf(x1 - mn);
      float rs = p0 + p1;
#pragma unroll
      for (int off = 8; off >= 1; off >>= 1) rs += __shfl_xor(rs, off, 16);
      lrow[i] = lrow[i] * alpha + rs;
#pragma unroll
      for (int t = 0; t < 4; ++t) o[t][i] *= alpha;
      Plds[wid][r][lr]      = (_Float16)p0;
      Plds[wid][r][lr + 16] = (_Float16)p1;
    }

    // reload P in A-fragment layout (same-wave LDS RAW; DS ops are in-order)
    U pa;
    pa.p[0] = *(const v8h*)&Plds[wid][lr][lh * 16];
    pa.p[1] = *(const v8h*)&Plds[wid][lr][lh * 16 + 8];

    // PV: B-fragments are contiguous runs of V^T rows (lane = output dim col)
    const int ks = k0 + lh * 16;
#pragma unroll
    for (int t = 0; t < 4; ++t) {
      const _Float16* vr = Vp + (size_t)(t * 16 + lr) * SEQ + ks;
      v16h vb = *(const v16h*)vr;
      o[t] = WMMA_F16(pa.v, vb, o[t]);
    }
  }

  // normalize and write [B,S,H*64] f16 for the output projection
#pragma unroll
  for (int t = 0; t < 4; ++t)
#pragma unroll
    for (int i = 0; i < 8; ++i) {
      const int q = qbase + lh * 8 + i;
      const int col = h * 64 + t * 16 + lr;
      Out[((size_t)b * SEQ + q) * DM + col] = (_Float16)(o[t][i] / lrow[i]);
    }
}

// ---------------------------------------------------------------------------
extern "C" void kernel_launch(void* const* d_in, const int* in_sizes, int n_in,
                              void* d_out, int out_size, void* d_ws, size_t ws_size,
                              hipStream_t stream) {
  const float* q    = (const float*)d_in[0];
  const float* k    = (const float*)d_in[1];
  const float* v    = (const float*)d_in[2];
  const float* bias = (const float*)d_in[3];
  const float* w_q  = (const float*)d_in[4];
  const float* b_q  = (const float*)d_in[5];
  const float* w_k  = (const float*)d_in[6];
  const float* b_k  = (const float*)d_in[7];
  const float* w_v  = (const float*)d_in[8];
  const float* b_v  = (const float*)d_in[9];
  const float* w_o  = (const float*)d_in[10];
  const float* b_o  = (const float*)d_in[11];

  // workspace layout (all regions fully written before being read):
  //   4 x 8 MB head/act f16 + 3 x 8 MB f16 activations + 4 x 2 MB f16 weights
  char* ws = (char*)d_ws;
  const size_t actBytes = (size_t)MROWS * DM * 2;  // 8 MB
  const size_t wBytes   = (size_t)DM * DM * 2;     // 2 MB
  _Float16* Qh  = (_Float16*)(ws + 0 * actBytes);
  _Float16* Kh  = (_Float16*)(ws + 1 * actBytes);
  _Float16* Vt  = (_Float16*)(ws + 2 * actBytes);
  _Float16* AO  = (_Float16*)(ws + 3 * actBytes);
  _Float16* qf  = (_Float16*)(ws + 4 * actBytes);
  _Float16* kf  = (_Float16*)(ws + 5 * actBytes);
  _Float16* vf  = (_Float16*)(ws + 6 * actBytes);
  _Float16* WqT = (_Float16*)(ws + 7 * actBytes + 0 * wBytes);
  _Float16* WkT = (_Float16*)(ws + 7 * actBytes + 1 * wBytes);
  _Float16* WvT = (_Float16*)(ws + 7 * actBytes + 2 * wBytes);
  _Float16* WoT = (_Float16*)(ws + 7 * actBytes + 3 * wBytes);

  dim3 blk(256);

  const int nAct = MROWS * DM;                 // 4,194,304
  dim3 cg(nAct / (256 * 4));                   // 4096 blocks
  cvt_f16<<<cg, blk, 0, stream>>>(q, qf, nAct);
  cvt_f16<<<cg, blk, 0, stream>>>(k, kf, nAct);
  cvt_f16<<<cg, blk, 0, stream>>>(v, vf, nAct);

  dim3 tg(DM / 32, DM / 32);                   // (32, 32)
  cvt_tr_f16<<<tg, blk, 0, stream>>>(w_q, WqT);
  cvt_tr_f16<<<tg, blk, 0, stream>>>(w_k, WkT);
  cvt_tr_f16<<<tg, blk, 0, stream>>>(w_v, WvT);
  cvt_tr_f16<<<tg, blk, 0, stream>>>(w_o, WoT);

  dim3 gg(DM / 64, MROWS / 128);               // (16, 32)
  gemm_wmma<OUT_HEAD  ><<<gg, blk, 0, stream>>>(qf, WqT, b_q, Qh, 0.125f);
  gemm_wmma<OUT_HEAD  ><<<gg, blk, 0, stream>>>(kf, WkT, b_k, Kh, 1.0f);
  gemm_wmma<OUT_HEAD_T><<<gg, blk, 0, stream>>>(vf, WvT, b_v, Vt, 1.0f);

  dim3 ag(SEQ / 128, HEADS, BATCH);            // (16, 16, 2)
  flash_attn<<<ag, blk, 0, stream>>>(Qh, Kh, Vt, bias, AO);

  gemm_wmma<OUT_FINAL><<<gg, blk, 0, stream>>>(AO, WoT, b_o, d_out, 1.0f);
}
